// ONLSTMCell_8546984919639
// MI455X (gfx1250) — compile-verified, matrix operand order checked by reading, not run on hardware
//
#include <hip/hip_runtime.h>
#include <stdint.h>

// ---------------------------------------------------------------------------
// ON-LSTM cell, MI455X (gfx1250).
// bf16 WMMA dual GEMM (v_wmma_f32_16x16x32_bf16, f32 accumulate) with fused
// ON-LSTM gate epilogue. Staging uses CDNA5 async global->LDS
// (global_load_async_to_lds_b128, ASYNCcnt) with sync fallback.
// Tile: 128 batch rows x (4 gate groups x 64 cs cols) per block, K = 2*1024.
// Workspace (~49.5 MB):
//   [0,16MB)   x  bf16   [16,32MB) hx bf16
//   [32,40MB)  W_ih bf16 [40,48MB) W_hh bf16
//   [48MB..)   cingate f32 (8192*16), cforget f32 (8192*16)
// ---------------------------------------------------------------------------

typedef __attribute__((ext_vector_type(16))) __bf16 v16bf;
typedef __attribute__((ext_vector_type(8)))  float  v8f;
typedef int v4i __attribute__((vector_size(16)));
typedef __attribute__((address_space(1))) v4i* gv4i_p;
typedef __attribute__((address_space(3))) v4i* lv4i_p;

#define B_SZ   8192
#define IN_SZ  1024
#define H_SZ   1024
#define NCHUNK 16
#define CSZ    64
#define BM     128

#if __has_builtin(__builtin_amdgcn_global_load_async_to_lds_b128)
#define HAVE_ASYNC_LDS 1
#else
#define HAVE_ASYNC_LDS 0
#endif

// ---------------- fp32 -> bf16 (round-to-nearest-even) ----------------------
__device__ __forceinline__ unsigned short f32_to_bf16_rne(float f) {
  unsigned int u = __float_as_uint(f);
  unsigned int r = u + 0x7FFFu + ((u >> 16) & 1u);
  return (unsigned short)(r >> 16);
}

__global__ void cvt_bf16_kernel(const float4* __restrict__ src,
                                uint4* __restrict__ dst, int n8) {
  int i = blockIdx.x * blockDim.x + threadIdx.x;
  if (i >= n8) return;
  float4 a = src[2 * i + 0];
  float4 b = src[2 * i + 1];
  union { unsigned short h[8]; uint4 q; } u;
  u.h[0] = f32_to_bf16_rne(a.x); u.h[1] = f32_to_bf16_rne(a.y);
  u.h[2] = f32_to_bf16_rne(a.z); u.h[3] = f32_to_bf16_rne(a.w);
  u.h[4] = f32_to_bf16_rne(b.x); u.h[5] = f32_to_bf16_rne(b.y);
  u.h[6] = f32_to_bf16_rne(b.z); u.h[7] = f32_to_bf16_rne(b.w);
  dst[i] = u.q;
}

// ---------------- cumulative softmax over 8 elements ------------------------
__device__ __forceinline__ void cumsoftmax8_fwd(const float* __restrict__ x,
                                                float* __restrict__ o) {
  float m = x[0];
#pragma unroll
  for (int i = 1; i < 8; ++i) m = fmaxf(m, x[i]);
  float e[8], s = 0.f;
#pragma unroll
  for (int i = 0; i < 8; ++i) { e[i] = __expf(x[i] - m); s += e[i]; }
  float inv = __frcp_rn(s), c = 0.f;
#pragma unroll
  for (int i = 0; i < 8; ++i) { c += e[i] * inv; o[i] = c; }
}

__device__ __forceinline__ void cumsoftmax8_rev(const float* __restrict__ x,
                                                float* __restrict__ o) {
  float m = x[0];
#pragma unroll
  for (int i = 1; i < 8; ++i) m = fmaxf(m, x[i]);
  float e[8], s = 0.f;
#pragma unroll
  for (int i = 0; i < 8; ++i) { e[i] = __expf(x[i] - m); s += e[i]; }
  float inv = __frcp_rn(s), c = 0.f;
#pragma unroll
  for (int i = 7; i >= 0; --i) { c += e[i] * inv; o[i] = c; }
}

__global__ void gatecum_kernel(const float* __restrict__ infloor,
                               const float* __restrict__ fgfloor,
                               float* __restrict__ cing,
                               float* __restrict__ cforg) {
  int b = blockIdx.x * blockDim.x + threadIdx.x;
  if (b >= B_SZ) return;
  float inr[16], fgr[16], oc[16], of[16];
#pragma unroll
  for (int i = 0; i < 16; ++i) {
    inr[i] = infloor[b * 16 + i];
    fgr[i] = fgfloor[b * 16 + i];
  }
  cumsoftmax8_fwd(&inr[0], &oc[0]);  // happy_cingate
  cumsoftmax8_rev(&inr[8], &oc[8]);  // sad_cingate
  cumsoftmax8_rev(&fgr[0], &of[0]);  // happy_cforgetgate
  cumsoftmax8_fwd(&fgr[8], &of[8]);  // sad_cforgetgate
#pragma unroll
  for (int i = 0; i < 16; ++i) {
    cing[b * 16 + i]  = oc[i];
    cforg[b * 16 + i] = of[i];
  }
}

// ---------------- staging: CDNA5 async global->LDS (fallback: sync) ---------
__device__ __forceinline__ void stage16(void* lds_dst, const void* gsrc) {
#if HAVE_ASYNC_LDS
  __builtin_amdgcn_global_load_async_to_lds_b128(
      (gv4i_p)gsrc, (lv4i_p)lds_dst, 0, 0);
#else
  *(uint4*)lds_dst = *(const uint4*)gsrc;
#endif
}

__device__ __forceinline__ void stage_wait() {
#if HAVE_ASYNC_LDS
#if __has_builtin(__builtin_amdgcn_s_wait_asynccnt)
  __builtin_amdgcn_s_wait_asynccnt(0);
#else
  asm volatile("s_wait_asynccnt 0x0" ::: "memory");
#endif
#endif
}

// ---------------- WMMA fragment load (CDNA5 16-bit A/B layout) --------------
// lane<16 holds K {0..7,16..23}, lane>=16 holds K {8..15,24..31}:
// two contiguous 16B chunks per fragment.
__device__ __forceinline__ v16bf load_frag(const unsigned short* __restrict__ row,
                                           int ks, int halfk) {
  union { v16bf v; uint4 q[2]; } u;
  const int k0 = ks * 32 + halfk * 8;
  u.q[0] = *(const uint4*)(row + k0);
  u.q[1] = *(const uint4*)(row + k0 + 16);
  return u.v;
}

__device__ __forceinline__ float sigmoidf_fast(float x) {
  return 1.0f / (1.0f + __expf(-x));
}

// ---------------- main fused dual-GEMM + ON-LSTM epilogue -------------------
// Block: 256 threads (8 wave32). Tile: 128 batch rows x (4 groups x 64 cs).
// Wave w: cs-subtile q=w&3, row-quad rp=w>>2 (row tiles 4*rp..4*rp+3).
// Each wave owns all 4 gate groups for its 16 cs columns -> wave-local
// epilogue; per K=32 step: 4 A-frags + 4 B-frags -> 16 WMMAs.
__global__ __launch_bounds__(256)
void onlstm_wmma_kernel(const unsigned short* __restrict__ xb,
                        const unsigned short* __restrict__ hb,
                        const unsigned short* __restrict__ wih,
                        const unsigned short* __restrict__ whh,
                        const float* __restrict__ bih,
                        const float* __restrict__ bhh,
                        const float* __restrict__ cx,
                        const float* __restrict__ cing,
                        const float* __restrict__ cforg,
                        float* __restrict__ hy,
                        float* __restrict__ cyo) {
  __shared__ __align__(16) unsigned short lA[BM * 64];    // 16 KB
  __shared__ __align__(16) unsigned short lB[256 * 64];   // 32 KB

  const int m0   = blockIdx.x * BM;   // batch tile
  const int nc   = blockIdx.y;        // chunk index (0..15)
  const int tid  = threadIdx.x;
  const int lane = tid & 31;
  const int w    = tid >> 5;
  const int q    = w & 3;             // cs subtile (16 cols) within chunk
  const int rp   = w >> 2;            // row quad: row tiles 4*rp..4*rp+3
  const int hk   = lane >> 4;         // K-half within fragment
  const int ln   = lane & 15;         // matrix row/col within 16

  v8f acc[4][4];                      // [row-tile r][gate group g]
#pragma unroll
  for (int r = 0; r < 4; ++r)
#pragma unroll
    for (int g = 0; g < 4; ++g) acc[r][g] = 0.f;

  // 32 K-tiles of 64: t<16 -> x @ W_ih^T, t>=16 -> hx @ W_hh^T (same accums).
  for (int t = 0; t < 32; ++t) {
    const unsigned short* __restrict__ As = (t < 16) ? xb : hb;
    const unsigned short* __restrict__ Bs = (t < 16) ? wih : whh;
    const int k0 = (t & 15) * 64;

    // Stage A tile: 128 rows x 64 bf16 = 1024 x 16B (4 per thread).
#pragma unroll
    for (int i = 0; i < 4; ++i) {
      int u = tid + i * 256;
      int row = u >> 3, c = u & 7;
      stage16((char*)lA + (size_t)u * 16,
              As + (size_t)(m0 + row) * IN_SZ + k0 + c * 8);
    }
    // Stage B tile: 256 gate-cols x 64 bf16 = 2048 x 16B (8 per thread).
    // Gate column cc -> weight row j = (cc/64)*1024 + nc*64 + (cc%64).
#pragma unroll
    for (int i = 0; i < 8; ++i) {
      int u = tid + i * 256;
      int cc = u >> 3, c = u & 7;
      int j = (cc >> 6) * H_SZ + nc * CSZ + (cc & 63);
      stage16((char*)lB + (size_t)u * 16,
              Bs + (size_t)j * IN_SZ + k0 + c * 8);
    }
    // L2 prefetch of next K-tile (global_prefetch_b8).
    if (t < 31) {
      const unsigned short* Bsn = (t + 1 < 16) ? wih : whh;
      int k0n = ((t + 1) & 15) * 64;
      int jj = (tid >> 6) * H_SZ + nc * CSZ + (tid & 63);
      __builtin_prefetch(Bsn + (size_t)jj * IN_SZ + k0n, 0, 1);
    }
    stage_wait();
    __syncthreads();

#pragma unroll
    for (int ks = 0; ks < 2; ++ks) {
      v16bf af[4];
#pragma unroll
      for (int r = 0; r < 4; ++r)
        af[r] = load_frag(lA + ((rp * 4 + r) * 16 + ln) * 64, ks, hk);
#pragma unroll
      for (int g = 0; g < 4; ++g) {
        v16bf bf = load_frag(lB + ((g * 4 + q) * 16 + ln) * 64, ks, hk);
#pragma unroll
        for (int r = 0; r < 4; ++r) {
          acc[r][g] = __builtin_amdgcn_wmma_f32_16x16x32_bf16(
              false, af[r], false, bf, (short)0, acc[r][g], false, false);
        }
      }
    }
    __syncthreads();
  }

  // ----- fused epilogue: C/D layout => lane: N=ln, M = hk*8 + v ------------
  const int cs = q * 16 + ln;                         // 0..63 within chunk
  float bsum[4];
#pragma unroll
  for (int g = 0; g < 4; ++g) {
    const int j = g * H_SZ + nc * CSZ + cs;
    bsum[g] = bih[j] + bhh[j];
  }
#pragma unroll
  for (int r = 0; r < 4; ++r) {
#pragma unroll
    for (int v = 0; v < 8; ++v) {
      const int b = m0 + (rp * 4 + r) * 16 + hk * 8 + v;
      const float go   = sigmoidf_fast(acc[r][0][v] + bsum[0]); // outgate
      const float cell = tanhf(acc[r][1][v] + bsum[1]);         // cell
      const float ig   = sigmoidf_fast(acc[r][2][v] + bsum[2]); // ingate
      const float fg   = sigmoidf_fast(acc[r][3][v] + bsum[3]); // forgetgate
      const float ci = cing[b * NCHUNK + nc];
      const float cf = cforg[b * NCHUNK + nc];
      const float ov = cf * ci;
      const float fgate = fg * ov + (cf - ov);
      const float igate = ig * ov + (ci - ov);
      const size_t idx = (size_t)b * H_SZ + nc * CSZ + cs;
      const float cy = fgate * cx[idx] + igate * cell;
      hy[idx]  = go * tanhf(cy);
      cyo[idx] = cy;
    }
  }
}

// ---------------------------------------------------------------------------
extern "C" void kernel_launch(void* const* d_in, const int* in_sizes, int n_in,
                              void* d_out, int out_size, void* d_ws, size_t ws_size,
                              hipStream_t stream) {
  const float* x    = (const float*)d_in[0];
  const float* hx   = (const float*)d_in[1];
  const float* cx   = (const float*)d_in[2];
  const float* infl = (const float*)d_in[3];
  const float* fgfl = (const float*)d_in[4];
  const float* Wih  = (const float*)d_in[5];
  const float* bih  = (const float*)d_in[6];
  const float* Whh  = (const float*)d_in[7];
  const float* bhh  = (const float*)d_in[8];

  char* ws = (char*)d_ws;
  unsigned short* xb   = (unsigned short*)(ws);
  unsigned short* hb   = (unsigned short*)(ws + ((size_t)16 << 20));
  unsigned short* wihb = (unsigned short*)(ws + ((size_t)32 << 20));
  unsigned short* whhb = (unsigned short*)(ws + ((size_t)40 << 20));
  float* cing  = (float*)(ws + ((size_t)48 << 20));
  float* cforg = cing + (size_t)B_SZ * NCHUNK;

  float* hy  = (float*)d_out;
  float* cyo = hy + (size_t)B_SZ * H_SZ;

  // fp32 -> bf16 conversions (8 elements per thread)
  {
    int n8;
    n8 = (B_SZ * IN_SZ) / 8;
    cvt_bf16_kernel<<<(n8 + 255) / 256, 256, 0, stream>>>((const float4*)x,  (uint4*)xb,   n8);
    n8 = (B_SZ * H_SZ) / 8;
    cvt_bf16_kernel<<<(n8 + 255) / 256, 256, 0, stream>>>((const float4*)hx, (uint4*)hb,   n8);
    n8 = (4 * H_SZ * IN_SZ) / 8;
    cvt_bf16_kernel<<<(n8 + 255) / 256, 256, 0, stream>>>((const float4*)Wih, (uint4*)wihb, n8);
    n8 = (4 * H_SZ * H_SZ) / 8;
    cvt_bf16_kernel<<<(n8 + 255) / 256, 256, 0, stream>>>((const float4*)Whh, (uint4*)whhb, n8);
  }

  // master-gate cumulative softmaxes
  gatecum_kernel<<<B_SZ / 256, 256, 0, stream>>>(infl, fgfl, cing, cforg);

  // fused dual GEMM + ON-LSTM epilogue
  dim3 grid(B_SZ / BM, NCHUNK);
  onlstm_wmma_kernel<<<grid, 256, 0, stream>>>(xb, hb, wihb, whhb,
                                               bih, bhh, cx, cing, cforg,
                                               hy, cyo);
}
